// PosAttBiLSTM_67869073211902
// MI455X (gfx1250) — matrix operand order repaired and checked by reading, not compiled, not run on hardware
//
#include <hip/hip_runtime.h>
#include <math.h>
#include <stdint.h>

// ---------------- problem dims ----------------
#define S_LEN   1024
#define B_SZ    8
#define E_DIM   256
#define H_DIM   512
#define OUT_DIM 5
#define WIN_SZ  30
#define BS_TOT  (B_SZ * S_LEN)     // 8192
#define G4H     (4 * H_DIM)        // 2048

#ifndef USE_TDM
#define USE_TDM 1                  // flip to 0 for plain-load staging
#endif

// ---------------- CDNA5 types ----------------
typedef __attribute__((ext_vector_type(16))) __bf16 v16bf;
typedef __attribute__((ext_vector_type(8)))  float  v8f;

union Frag16 { v16bf v; unsigned u[8]; };

__device__ __forceinline__ unsigned short f2bf(float f) {
  unsigned u = __float_as_uint(f);
  u += 0x7FFFu + ((u >> 16) & 1u);       // round-to-nearest-even
  return (unsigned short)(u >> 16);
}

__device__ __forceinline__ v8f wmma_bf16(const Frag16& a, const Frag16& b, v8f c) {
  return __builtin_amdgcn_wmma_f32_16x16x32_bf16(false, a.v, false, b.v, (short)0, c, false, false);
}

#if USE_TDM
typedef unsigned int tdm_v4u __attribute__((ext_vector_type(4)));
typedef int          tdm_v8i __attribute__((ext_vector_type(8)));
typedef int          tdm_v4i __attribute__((ext_vector_type(4)));

// TDM load of a rows x 32-bf16 tile; LDS row stride padded to 40 bf16
// (pad_interval = 16 dwords, pad_amount = 4 dwords) per CDNA5 ISA D#.
__device__ __forceinline__ void tdm_load_tile(unsigned lds_byte_off, const void* gaddr,
                                              unsigned rows, unsigned strideK_elems) {
  unsigned long long ga = (unsigned long long)(uintptr_t)gaddr;
  const unsigned tdim0 = 32u, tile0 = 32u;
  tdm_v4u g0;
  g0[0] = 1u;                                        // count=1, user mode
  g0[1] = lds_byte_off;                              // lds_addr (bytes)
  g0[2] = (unsigned)(ga & 0xFFFFFFFFull);            // global_addr[31:0]
  g0[3] = (unsigned)((ga >> 32) & 0x1FFFFFFull) | (2u << 30);  // global_addr[56:32], type=2
  tdm_v8i g1;
  g1[0] = (int)((1u << 16)        // data_size = 2 bytes
              | (1u << 20)        // pad_enable
              | (3u << 22)        // pad_interval: 16 dwords
              | (3u << 25));      // pad_amount: 4 dwords -> LDS row stride 40 elems
  g1[1] = (int)((tdim0 & 0xFFFFu) << 16);                      // tensor_dim0[15:0]
  g1[2] = (int)((tdim0 >> 16) | ((rows & 0xFFFFu) << 16));     // dim0[31:16], dim1[15:0]
  g1[3] = (int)((rows >> 16) | (tile0 << 16));                 // dim1[31:16], tile_dim0
  g1[4] = (int)(rows & 0xFFFFu);                               // tile_dim1 (tile_dim2=0)
  g1[5] = (int)strideK_elems;                                  // tensor_dim0_stride[31:0]
  g1[6] = 0;
  g1[7] = 0;
  tdm_v4i g2 = {0, 0, 0, 0};
  tdm_v4i g3 = {0, 0, 0, 0};
#if defined(__clang_major__) && (__clang_major__ >= 23)
  tdm_v8i g4 = {0, 0, 0, 0, 0, 0, 0, 0};
  __builtin_amdgcn_tensor_load_to_lds(g0, g1, g2, g3, g4, 0);
#else
  __builtin_amdgcn_tensor_load_to_lds(g0, g1, g2, g3, 0);
#endif
}
#endif // USE_TDM

// ---------------- embedding + positional encoding -> bf16 ----------------
__global__ __launch_bounds__(256)
void k_embed(const int* __restrict__ text, const float* __restrict__ emb,
             unsigned short* __restrict__ xb) {
  int bs = blockIdx.x;
  int e  = threadIdx.x;
  int tok = text[bs];
  float v = emb[(size_t)tok * E_DIM + e];
  int s = bs & (S_LEN - 1);
  float div = __expf(-(float)(e & ~1) * (9.210340371976184f / 256.0f)); // ln(1e4)/E
  float ang = (float)s * div;
  float pe  = 0.5f * (((e & 1) ? __cosf(ang) : __sinf(ang)) + 1.0f);
  xb[(size_t)bs * E_DIM + e] = f2bf(v + pe);
}

// ---------------- fp32 -> bf16 convert ----------------
__global__ void k_cvt(const float* __restrict__ src, unsigned short* __restrict__ dst,
                      long long n) {
  for (long long i = blockIdx.x * (long long)blockDim.x + threadIdx.x; i < n;
       i += (long long)gridDim.x * blockDim.x)
    dst[i] = f2bf(src[i]);
}

// ---------------- generic WMMA GEMM: C = alpha*(A[MxK] * W[NxK]^T) + bias0 + bias1 -------
// 128x64 tile / WG (256 thr, 8 waves; each wave 32x32 = 4 x v_wmma_f32_16x16x32_bf16).
// Both tiles staged by the Tensor Data Mover into double-buffered LDS (row stride 40),
// prefetch of step k+1 issued right after the single tile-ready barrier of step k, so the
// DMA overlaps the WMMAs. No plain global loads in the hot loop.
__global__ __launch_bounds__(256)
void k_gemm(const unsigned short* __restrict__ A, const unsigned short* __restrict__ W,
            const float* __restrict__ bias0, const float* __restrict__ bias1,
            float* __restrict__ Cf, unsigned short* __restrict__ Cbf,
            int M, int N, int K, float alpha,
            long long strideA, long long strideW, long long strideC) {
  __shared__ unsigned short lsA[2 * 128 * 40];
  __shared__ unsigned short lsW[2 * 64 * 40];
  const int bz = blockIdx.z;
  A += (size_t)bz * strideA;
  W += (size_t)bz * strideW;
  const long long cbase = (long long)bz * strideC;
  const int m0 = blockIdx.y * 128;
  const int n0 = blockIdx.x * 64;
  const int tid = threadIdx.x;
  const int lane = tid & 31, wave = tid >> 5;
  const int wm = (wave >> 1) * 32;     // wave rows [wm, wm+32)
  const int wn = (wave & 1) * 32;      // wave cols [wn, wn+32)
  v8f acc00 = {0, 0, 0, 0, 0, 0, 0, 0};
  v8f acc01 = {0, 0, 0, 0, 0, 0, 0, 0};
  v8f acc10 = {0, 0, 0, 0, 0, 0, 0, 0};
  v8f acc11 = {0, 0, 0, 0, 0, 0, 0, 0};
  // A-fragment lane mapping (ISA 16-bit A 16x32 layout)
  const int am  = lane & 15;
  const int akb = (lane < 16) ? 0 : 8;
  // B-fragment lane mapping (16-bit B 32x16: lanes 0-15 K=0..7,16..23, lanes 16-31 K=8..15,24..31)
  const int bn  = lane & 15;
  const int bkb = (lane < 16) ? 0 : 8;

#if USE_TDM
  if (wave == 0) {                      // prologue: stage k0 = 0 into buffer 0
    tdm_load_tile((unsigned)(size_t)&lsA[0], A + (size_t)m0 * K, 128u, (unsigned)K);
    tdm_load_tile((unsigned)(size_t)&lsW[0], W + (size_t)n0 * K, 64u, (unsigned)K);
  }
#endif
  int buf = 0;
  for (int k0 = 0; k0 < K; k0 += 32) {
    const unsigned short* As = &lsA[buf * (128 * 40)];
    const unsigned short* Ws = &lsW[buf * (64 * 40)];
#if USE_TDM
    if (wave == 0) __builtin_amdgcn_s_wait_tensorcnt(0);
    __syncthreads();                    // tile ready; all reads of the other buffer done
    if (wave == 0 && (k0 + 32 < K)) {   // prefetch next tile, overlapped with WMMAs below
      unsigned nb = (unsigned)(buf ^ 1);
      tdm_load_tile((unsigned)(size_t)&lsA[nb * (128 * 40)],
                    A + (size_t)m0 * K + k0 + 32, 128u, (unsigned)K);
      tdm_load_tile((unsigned)(size_t)&lsW[nb * (64 * 40)],
                    W + (size_t)n0 * K + k0 + 32, 64u, (unsigned)K);
    }
#else
    {
      int r = tid >> 1, c = (tid & 1) * 16;   // A tile: 128 x 32, 16 elems / thread
      *(uint4*)&lsA[buf * (128 * 40) + r * 40 + c] =
          *(const uint4*)(A + (size_t)(m0 + r) * K + k0 + c);
      *(uint4*)&lsA[buf * (128 * 40) + r * 40 + c + 8] =
          *(const uint4*)(A + (size_t)(m0 + r) * K + k0 + c + 8);
      int rw = tid >> 2, cw = (tid & 3) * 8;  // W tile: 64 x 32, 8 elems / thread
      *(uint4*)&lsW[buf * (64 * 40) + rw * 40 + cw] =
          *(const uint4*)(W + (size_t)(n0 + rw) * K + k0 + cw);
    }
    __syncthreads();
#endif
    Frag16 a0, a1, b0f, b1f;
    *(uint4*)&a0.u[0]  = *(const uint4*)&As[(wm + am) * 40 + akb];
    *(uint4*)&a0.u[4]  = *(const uint4*)&As[(wm + am) * 40 + akb + 16];
    *(uint4*)&a1.u[0]  = *(const uint4*)&As[(wm + 16 + am) * 40 + akb];
    *(uint4*)&a1.u[4]  = *(const uint4*)&As[(wm + 16 + am) * 40 + akb + 16];
    *(uint4*)&b0f.u[0] = *(const uint4*)&Ws[(wn + bn) * 40 + bkb];
    *(uint4*)&b0f.u[4] = *(const uint4*)&Ws[(wn + bn) * 40 + bkb + 16];
    *(uint4*)&b1f.u[0] = *(const uint4*)&Ws[(wn + 16 + bn) * 40 + bkb];
    *(uint4*)&b1f.u[4] = *(const uint4*)&Ws[(wn + 16 + bn) * 40 + bkb + 16];
    acc00 = wmma_bf16(a0, b0f, acc00);
    acc01 = wmma_bf16(a0, b1f, acc01);
    acc10 = wmma_bf16(a1, b0f, acc10);
    acc11 = wmma_bf16(a1, b1f, acc11);
    buf ^= 1;
  }
  // C/D layout: lane col = lane&15, row = r + (lane>=16 ? 8 : 0)
  const int cn = lane & 15;
  const int radd = (lane < 16) ? 0 : 8;
  const int nA = n0 + wn + cn;
  const int nB = nA + 16;
  const float bA = (bias0 ? bias0[nA] : 0.f) + (bias1 ? bias1[nA] : 0.f);
  const float bB = (bias0 ? bias0[nB] : 0.f) + (bias1 ? bias1[nB] : 0.f);
#pragma unroll
  for (int r = 0; r < 8; ++r) {
    int mA = m0 + wm + radd + r;
    int mB = mA + 16;
    float v00 = acc00[r] * alpha + bA;
    float v01 = acc01[r] * alpha + bB;
    float v10 = acc10[r] * alpha + bA;
    float v11 = acc11[r] * alpha + bB;
    long long i00 = cbase + (long long)mA * N + nA;
    long long i01 = cbase + (long long)mA * N + nB;
    long long i10 = cbase + (long long)mB * N + nA;
    long long i11 = cbase + (long long)mB * N + nB;
    if (Cf)  { Cf[i00] = v00; Cf[i01] = v01; Cf[i10] = v10; Cf[i11] = v11; }
    if (Cbf) { Cbf[i00] = f2bf(v00); Cbf[i01] = f2bf(v01);
               Cbf[i10] = f2bf(v10); Cbf[i11] = f2bf(v11); }
  }
}

// ---------------- BiLSTM recurrence ----------------
// grid(8,2): 8 WGs per direction, each owns 64 hidden units (256 gate columns).
// h (8x512) exchanged through L2 each step; w_hh stays L2-resident (b128 streams);
// c stays in registers. Inter-WG sync: monotonic atomic counter (16 WGs co-resident).
__global__ __launch_bounds__(256)
void k_lstm(const float* __restrict__ xgF, const float* __restrict__ xgB,
            const unsigned short* __restrict__ whhF, const unsigned short* __restrict__ whhB,
            float* __restrict__ hF, float* __restrict__ hB,
            float* __restrict__ hstate, unsigned* __restrict__ sync) {
  const int NWG = 8;
  const int wg   = blockIdx.x;
  const int dir  = blockIdx.y;
  const int tid  = threadIdx.x;
  const int lane = tid & 31, wave = tid >> 5;
  const int hid0 = wg * 64;
  const float* xg           = dir ? xgB : xgF;
  const unsigned short* whh = dir ? whhB : whhF;
  float* hseq               = dir ? hB : hF;
  float* hbuf = hstate + (size_t)dir * 2 * (B_SZ * H_DIM);
  unsigned* cnt = sync + dir;

  __shared__ unsigned short hA[16 * 520];   // 16x512 bf16, stride 520
  __shared__ float gbuf[256 * 8];           // [gate_col][batch]

  const int am  = lane & 15;
  const int akb = (lane < 16) ? 0 : 8;
  const int bn  = lane & 15;
  const int bkb = (lane < 16) ? 0 : 8;
  const int n0  = wave * 32;
  const int nT0 = n0 + bn,      nT1 = n0 + 16 + bn;
  const int rowA = ((nT0 >> 6) * H_DIM) + hid0 + (nT0 & 63);
  const int rowB = ((nT1 >> 6) * H_DIM) + hid0 + (nT1 & 63);
  const unsigned short* wRowA = whh + (size_t)rowA * H_DIM + bkb;
  const unsigned short* wRowB = whh + (size_t)rowB * H_DIM + bkb;

  float creg[2] = {0.f, 0.f};
  int buf = 0;

  for (int t = 0; t < S_LEN; ++t) {
    const int s = dir ? (S_LEN - 1 - t) : t;
    const float* hprev = hbuf + buf * (B_SZ * H_DIM);
    float* hnext       = hbuf + (buf ^ 1) * (B_SZ * H_DIM);

    // stage h_prev into LDS as bf16, zero-pad rows 8..15
    {
      int row = tid >> 4;
      int c0  = (tid & 15) * 32;
      if (row < 8) {
#pragma unroll 8
        for (int j = 0; j < 32; ++j)
          hA[row * 520 + c0 + j] = f2bf(hprev[row * H_DIM + c0 + j]);
      } else {
#pragma unroll 8
        for (int j = 0; j < 32; ++j) hA[row * 520 + c0 + j] = 0;
      }
    }
    __syncthreads();

    // gates = h_prev @ w_hh^T  (M=16 pad, N=32 per wave, K=512)
    v8f acc0 = {0, 0, 0, 0, 0, 0, 0, 0};
    v8f acc1 = {0, 0, 0, 0, 0, 0, 0, 0};
    for (int kk = 0; kk < H_DIM; kk += 32) {
      Frag16 a, w0, w1;
      *(uint4*)&a.u[0]  = *(const uint4*)&hA[am * 520 + kk + akb];
      *(uint4*)&a.u[4]  = *(const uint4*)&hA[am * 520 + kk + akb + 16];
      *(uint4*)&w0.u[0] = *(const uint4*)(wRowA + kk);       // b128 streams from L2
      *(uint4*)&w0.u[4] = *(const uint4*)(wRowA + kk + 16);
      *(uint4*)&w1.u[0] = *(const uint4*)(wRowB + kk);
      *(uint4*)&w1.u[4] = *(const uint4*)(wRowB + kk + 16);
      acc0 = wmma_bf16(a, w0, acc0);
      acc1 = wmma_bf16(a, w1, acc1);
    }
    // scatter gates + xg into LDS
    {
      int cn = lane & 15;
      int radd = (lane < 16) ? 0 : 8;
#pragma unroll
      for (int r = 0; r < 8; ++r) {
        int m = radd + r;
        if (m < 8) {
          int nc0 = n0 + cn, nc1 = n0 + 16 + cn;
          int g0c = ((nc0 >> 6) * H_DIM) + hid0 + (nc0 & 63);
          int g1c = ((nc1 >> 6) * H_DIM) + hid0 + (nc1 & 63);
          size_t xrow = ((size_t)m * S_LEN + s) * G4H;
          gbuf[nc0 * 8 + m] = acc0[r] + xg[xrow + g0c];
          gbuf[nc1 * 8 + m] = acc1[r] + xg[xrow + g1c];
        }
      }
    }
    __syncthreads();
    // LSTM cell pointwise: 2 (hidden,batch) pairs per thread
#pragma unroll
    for (int e = 0; e < 2; ++e) {
      int p = tid * 2 + e;
      int jj = p >> 3, b = p & 7;
      float gi = gbuf[(      jj) * 8 + b];
      float gf = gbuf[( 64 + jj) * 8 + b];
      float gg = gbuf[(128 + jj) * 8 + b];
      float go = gbuf[(192 + jj) * 8 + b];
      float si = 1.f / (1.f + __expf(-gi));
      float sf = 1.f / (1.f + __expf(-gf));
      float so = 1.f / (1.f + __expf(-go));
      float cv = sf * creg[e] + si * tanhf(gg);
      creg[e] = cv;
      float hv = so * tanhf(cv);
      hnext[b * H_DIM + hid0 + jj] = hv;
      hseq[((size_t)b * S_LEN + s) * H_DIM + hid0 + jj] = hv;
    }
    // inter-WG barrier (per direction)
    __syncthreads();
    if (tid == 0) {
      __threadfence();
      atomicAdd(cnt, 1u);
      unsigned tgt = (unsigned)(NWG * (t + 1));
      while (atomicAdd(cnt, 0u) < tgt) { __builtin_amdgcn_s_sleep(2); }
      __threadfence();
    }
    __syncthreads();
    buf ^= 1;
  }
}

// ---------------- concat hf,hb -> bf16 ----------------
__global__ void k_concat(const float* __restrict__ hf, const float* __restrict__ hb,
                         unsigned short* __restrict__ out) {
  long long n = (long long)BS_TOT * 1024;
  for (long long i = blockIdx.x * (long long)blockDim.x + threadIdx.x; i < n;
       i += (long long)gridDim.x * blockDim.x) {
    long long bs = i >> 10;
    int d = (int)(i & 1023);
    float v = (d < H_DIM) ? hf[bs * H_DIM + d] : hb[bs * H_DIM + d - H_DIM];
    out[i] = f2bf(v);
  }
}

// ---------------- gate = sigmoid(h . Wg + bg) ----------------
__global__ __launch_bounds__(256)
void k_gate(const float* __restrict__ h, const float* __restrict__ Wg,
            const float* __restrict__ bg, float* __restrict__ gate) {
  __shared__ float red[256];
  int bs = blockIdx.x, tid = threadIdx.x;
  float s = h[(size_t)bs * H_DIM + tid] * Wg[tid] +
            h[(size_t)bs * H_DIM + 256 + tid] * Wg[256 + tid];
  red[tid] = s; __syncthreads();
  for (int st = 128; st > 0; st >>= 1) {
    if (tid < st) red[tid] += red[tid + st];
    __syncthreads();
  }
  if (tid == 0) gate[bs] = 1.f / (1.f + __expf(-(red[0] + bg[0])));
}

// ---------------- V transpose: Vt[b][d][s] = V[b][s][d] ----------------
__global__ __launch_bounds__(256)
void k_transp(const unsigned short* __restrict__ Vb, unsigned short* __restrict__ Vt) {
  __shared__ unsigned short tt[32][33];
  int s0 = blockIdx.x * 32, d0 = blockIdx.y * 32, b = blockIdx.z;
  int tx = threadIdx.x, ty = threadIdx.y;
#pragma unroll
  for (int i = 0; i < 4; ++i)
    tt[ty + 8 * i][tx] = Vb[((size_t)b * S_LEN + s0 + ty + 8 * i) * H_DIM + d0 + tx];
  __syncthreads();
#pragma unroll
  for (int i = 0; i < 4; ++i)
    Vt[((size_t)b * H_DIM + d0 + ty + 8 * i) * S_LEN + s0 + tx] = tt[tx][ty + 8 * i];
}

// ---------------- dual softmax (global + windowed) -> bf16 probs ----------------
__global__ __launch_bounds__(256)
void k_softmax(const float* __restrict__ scores, unsigned short* __restrict__ pg,
               unsigned short* __restrict__ pl) {
  __shared__ float red[256];
  int q = blockIdx.x, b = blockIdx.y, tid = threadIdx.x;
  const float* row = scores + ((size_t)b * S_LEN + q) * S_LEN;
  float x[4];
#pragma unroll
  for (int i = 0; i < 4; ++i) x[i] = row[tid + 256 * i];
  // global softmax
  float mx = fmaxf(fmaxf(x[0], x[1]), fmaxf(x[2], x[3]));
  red[tid] = mx; __syncthreads();
  for (int st = 128; st > 0; st >>= 1) { if (tid < st) red[tid] = fmaxf(red[tid], red[tid + st]); __syncthreads(); }
  mx = red[0]; __syncthreads();
  float e[4], sm = 0.f;
#pragma unroll
  for (int i = 0; i < 4; ++i) { e[i] = __expf(x[i] - mx); sm += e[i]; }
  red[tid] = sm; __syncthreads();
  for (int st = 128; st > 0; st >>= 1) { if (tid < st) red[tid] += red[tid + st]; __syncthreads(); }
  sm = red[0]; __syncthreads();
  float inv = 1.f / sm;
  unsigned short* og = pg + ((size_t)b * S_LEN + q) * S_LEN;
#pragma unroll
  for (int i = 0; i < 4; ++i) og[tid + 256 * i] = f2bf(e[i] * inv);
  // local (|k-q| <= WIN) softmax; zeros outside window
  bool valid[4];
  float mxl = -3.4e38f;
#pragma unroll
  for (int i = 0; i < 4; ++i) {
    int k = tid + 256 * i;
    int d = k - q; if (d < 0) d = -d;
    valid[i] = (d <= WIN_SZ);
    if (valid[i]) mxl = fmaxf(mxl, x[i]);
  }
  red[tid] = mxl; __syncthreads();
  for (int st = 128; st > 0; st >>= 1) { if (tid < st) red[tid] = fmaxf(red[tid], red[tid + st]); __syncthreads(); }
  mxl = red[0]; __syncthreads();
  float el[4], sml = 0.f;
#pragma unroll
  for (int i = 0; i < 4; ++i) { el[i] = valid[i] ? __expf(x[i] - mxl) : 0.f; sml += el[i]; }
  red[tid] = sml; __syncthreads();
  for (int st = 128; st > 0; st >>= 1) { if (tid < st) red[tid] += red[tid + st]; __syncthreads(); }
  sml = red[0]; __syncthreads();
  float invl = 1.f / sml;
  unsigned short* ol = pl + ((size_t)b * S_LEN + q) * S_LEN;
#pragma unroll
  for (int i = 0; i < 4; ++i) ol[tid + 256 * i] = f2bf(el[i] * invl);
}

// ---------------- attended = (1-gate)*outG + gate*outL ----------------
__global__ void k_combine(const float* __restrict__ og, const float* __restrict__ ol,
                          const float* __restrict__ gate, float* __restrict__ att) {
  long long n = (long long)BS_TOT * H_DIM;
  for (long long i = blockIdx.x * (long long)blockDim.x + threadIdx.x; i < n;
       i += (long long)gridDim.x * blockDim.x) {
    long long bs = i >> 9;
    float g = gate[bs];
    att[i] = (1.f - g) * og[i] + g * ol[i];
  }
}

// ---------------- pooled = [max_s, mean_s] ----------------
__global__ __launch_bounds__(512)
void k_pool(const float* __restrict__ att, float* __restrict__ pooled) {
  int b = blockIdx.x, d = threadIdx.x;
  float mx = -3.4e38f, sm = 0.f;
  for (int s = 0; s < S_LEN; ++s) {
    float v = att[((size_t)b * S_LEN + s) * H_DIM + d];
    mx = fmaxf(mx, v); sm += v;
  }
  pooled[b * 1024 + d] = mx;
  pooled[b * 1024 + H_DIM + d] = sm * (1.f / S_LEN);
}

// ---------------- batchnorm(B axis) + final FC ----------------
__global__ __launch_bounds__(1024)
void k_bnfc(const float* __restrict__ pooled, const float* __restrict__ g,
            const float* __restrict__ be, const float* __restrict__ Wfc,
            const float* __restrict__ bfc, float* __restrict__ out) {
  __shared__ float pn[B_SZ * 1024];
  int f = threadIdx.x;
  float mu = 0.f;
  for (int b = 0; b < B_SZ; ++b) mu += pooled[b * 1024 + f];
  mu *= 0.125f;
  float var = 0.f;
  for (int b = 0; b < B_SZ; ++b) { float d = pooled[b * 1024 + f] - mu; var += d * d; }
  var *= 0.125f;
  float inv = rsqrtf(var + 1e-5f);
  float gg = g[f], bb = be[f];
  for (int b = 0; b < B_SZ; ++b)
    pn[b * 1024 + f] = gg * (pooled[b * 1024 + f] - mu) * inv + bb;
  __syncthreads();
  if (f < B_SZ * OUT_DIM) {
    int b = f / OUT_DIM, o = f % OUT_DIM;
    float s = bfc[o];
    for (int k = 0; k < 1024; ++k) s += pn[b * 1024 + k] * Wfc[o * 1024 + k];
    out[b * OUT_DIM + o] = s;
  }
}

// ---------------- host ----------------
extern "C" void kernel_launch(void* const* d_in, const int* in_sizes, int n_in,
                              void* d_out, int out_size, void* d_ws, size_t ws_size,
                              hipStream_t stream) {
  (void)in_sizes; (void)n_in; (void)out_size; (void)ws_size;
  const int*   text   = (const int*)  d_in[0];
  const float* emb    = (const float*)d_in[1];
  const float* w_ih_f = (const float*)d_in[2];
  const float* w_hh_f = (const float*)d_in[3];
  const float* b_ih_f = (const float*)d_in[4];
  const float* b_hh_f = (const float*)d_in[5];
  const float* w_ih_b = (const float*)d_in[6];
  const float* w_hh_b = (const float*)d_in[7];
  const float* b_ih_b = (const float*)d_in[8];
  const float* b_hh_b = (const float*)d_in[9];
  const float* Wr  = (const float*)d_in[10];
  const float* br  = (const float*)d_in[11];
  const float* Wq  = (const float*)d_in[12];
  const float* bq  = (const float*)d_in[13];
  const float* Wk  = (const float*)d_in[14];
  const float* bk  = (const float*)d_in[15];
  const float* Wv  = (const float*)d_in[16];
  const float* bv  = (const float*)d_in[17];
  const float* Wg  = (const float*)d_in[18];
  const float* bg  = (const float*)d_in[19];
  const float* bng = (const float*)d_in[20];
  const float* bnb = (const float*)d_in[21];
  const float* Wfc = (const float*)d_in[22];
  const float* bfc = (const float*)d_in[23];

  char* w = (char*)d_ws;
  size_t off = 0;
  auto alloc = [&](size_t bytes) -> void* {
    void* p = w + off;
    off = (off + bytes + 255) & ~(size_t)255;
    return p;
  };
  unsigned short* xb     = (unsigned short*)alloc((size_t)BS_TOT * E_DIM * 2);
  unsigned short* wihf   = (unsigned short*)alloc((size_t)G4H * E_DIM * 2);
  unsigned short* whhf   = (unsigned short*)alloc((size_t)G4H * H_DIM * 2);
  unsigned short* wihb   = (unsigned short*)alloc((size_t)G4H * E_DIM * 2);
  unsigned short* whhb   = (unsigned short*)alloc((size_t)G4H * H_DIM * 2);
  unsigned short* wrb    = (unsigned short*)alloc((size_t)H_DIM * 1024 * 2);
  unsigned short* wqb    = (unsigned short*)alloc((size_t)H_DIM * H_DIM * 2);
  unsigned short* wkb    = (unsigned short*)alloc((size_t)H_DIM * H_DIM * 2);
  unsigned short* wvb    = (unsigned short*)alloc((size_t)H_DIM * H_DIM * 2);
  float* xgf    = (float*)alloc((size_t)BS_TOT * G4H * 4);
  float* xgb    = (float*)alloc((size_t)BS_TOT * G4H * 4);
  float* hf     = (float*)alloc((size_t)BS_TOT * H_DIM * 4);
  float* hb     = (float*)alloc((size_t)BS_TOT * H_DIM * 4);
  float* hstate = (float*)alloc((size_t)2 * 2 * B_SZ * H_DIM * 4);
  unsigned* syn = (unsigned*)alloc(256);
  unsigned short* outcat = (unsigned short*)alloc((size_t)BS_TOT * 1024 * 2);
  float* hproj           = (float*)alloc((size_t)BS_TOT * H_DIM * 4);
  unsigned short* hprojb = (unsigned short*)alloc((size_t)BS_TOT * H_DIM * 2);
  unsigned short* Qb = (unsigned short*)alloc((size_t)BS_TOT * H_DIM * 2);
  unsigned short* Kb = (unsigned short*)alloc((size_t)BS_TOT * H_DIM * 2);
  unsigned short* Vb = (unsigned short*)alloc((size_t)BS_TOT * H_DIM * 2);
  unsigned short* Vt = (unsigned short*)alloc((size_t)BS_TOT * H_DIM * 2);
  float* gate   = (float*)alloc((size_t)BS_TOT * 4);
  float* scores = (float*)alloc((size_t)B_SZ * S_LEN * S_LEN * 4);
  unsigned short* pgm = (unsigned short*)alloc((size_t)B_SZ * S_LEN * S_LEN * 2);
  unsigned short* plm = (unsigned short*)alloc((size_t)B_SZ * S_LEN * S_LEN * 2);
  float* outG   = (float*)alloc((size_t)BS_TOT * H_DIM * 4);
  float* outL   = (float*)alloc((size_t)BS_TOT * H_DIM * 4);
  float* att    = (float*)alloc((size_t)BS_TOT * H_DIM * 4);
  float* pooled = (float*)alloc((size_t)B_SZ * 1024 * 4);

  hipMemsetAsync(hstate, 0, (size_t)2 * 2 * B_SZ * H_DIM * 4, stream);
  hipMemsetAsync(syn, 0, 256, stream);

  k_embed<<<BS_TOT, 256, 0, stream>>>(text, emb, xb);
  k_cvt<<<512, 256, 0, stream>>>(w_ih_f, wihf, (long long)G4H * E_DIM);
  k_cvt<<<512, 256, 0, stream>>>(w_hh_f, whhf, (long long)G4H * H_DIM);
  k_cvt<<<512, 256, 0, stream>>>(w_ih_b, wihb, (long long)G4H * E_DIM);
  k_cvt<<<512, 256, 0, stream>>>(w_hh_b, whhb, (long long)G4H * H_DIM);
  k_cvt<<<512, 256, 0, stream>>>(Wr, wrb, (long long)H_DIM * 1024);
  k_cvt<<<512, 256, 0, stream>>>(Wq, wqb, (long long)H_DIM * H_DIM);
  k_cvt<<<512, 256, 0, stream>>>(Wk, wkb, (long long)H_DIM * H_DIM);
  k_cvt<<<512, 256, 0, stream>>>(Wv, wvb, (long long)H_DIM * H_DIM);

  // xg = x @ W_ih^T + b_ih + b_hh   (both directions)
  k_gemm<<<dim3(G4H / 64, BS_TOT / 128, 1), 256, 0, stream>>>(
      xb, wihf, b_ih_f, b_hh_f, xgf, nullptr, BS_TOT, G4H, E_DIM, 1.f, 0, 0, 0);
  k_gemm<<<dim3(G4H / 64, BS_TOT / 128, 1), 256, 0, stream>>>(
      xb, wihb, b_ih_b, b_hh_b, xgb, nullptr, BS_TOT, G4H, E_DIM, 1.f, 0, 0, 0);

  k_lstm<<<dim3(8, 2), 256, 0, stream>>>(xgf, xgb, whhf, whhb, hf, hb, hstate, syn);

  k_concat<<<4096, 256, 0, stream>>>(hf, hb, outcat);

  // h = out @ Wr^T + br  (f32 + bf16 copies)
  k_gemm<<<dim3(H_DIM / 64, BS_TOT / 128, 1), 256, 0, stream>>>(
      outcat, wrb, br, nullptr, hproj, hprojb, BS_TOT, H_DIM, 1024, 1.f, 0, 0, 0);

  // Q, K, V projections (bf16 out)
  k_gemm<<<dim3(H_DIM / 64, BS_TOT / 128, 1), 256, 0, stream>>>(
      hprojb, wqb, bq, nullptr, nullptr, Qb, BS_TOT, H_DIM, H_DIM, 1.f, 0, 0, 0);
  k_gemm<<<dim3(H_DIM / 64, BS_TOT / 128, 1), 256, 0, stream>>>(
      hprojb, wkb, bk, nullptr, nullptr, Kb, BS_TOT, H_DIM, H_DIM, 1.f, 0, 0, 0);
  k_gemm<<<dim3(H_DIM / 64, BS_TOT / 128, 1), 256, 0, stream>>>(
      hprojb, wvb, bv, nullptr, nullptr, Vb, BS_TOT, H_DIM, H_DIM, 1.f, 0, 0, 0);

  k_gate<<<BS_TOT, 256, 0, stream>>>(hproj, Wg, bg, gate);
  k_transp<<<dim3(S_LEN / 32, H_DIM / 32, B_SZ), dim3(32, 8), 0, stream>>>(Vb, Vt);

  // scores = Q K^T / sqrt(H), batched over B
  k_gemm<<<dim3(S_LEN / 64, S_LEN / 128, B_SZ), 256, 0, stream>>>(
      Qb, Kb, nullptr, nullptr, scores, nullptr, S_LEN, S_LEN, H_DIM,
      0.04419417382415922f, (long long)S_LEN * H_DIM, (long long)S_LEN * H_DIM,
      (long long)S_LEN * S_LEN);

  k_softmax<<<dim3(S_LEN, B_SZ), 256, 0, stream>>>(scores, pgm, plm);

  // attention outputs: probs @ V  (V pre-transposed), batched
  k_gemm<<<dim3(H_DIM / 64, S_LEN / 128, B_SZ), 256, 0, stream>>>(
      pgm, Vt, nullptr, nullptr, outG, nullptr, S_LEN, H_DIM, S_LEN, 1.f,
      (long long)S_LEN * S_LEN, (long long)H_DIM * S_LEN, (long long)S_LEN * H_DIM);
  k_gemm<<<dim3(H_DIM / 64, S_LEN / 128, B_SZ), 256, 0, stream>>>(
      plm, Vt, nullptr, nullptr, outL, nullptr, S_LEN, H_DIM, S_LEN, 1.f,
      (long long)S_LEN * S_LEN, (long long)H_DIM * S_LEN, (long long)S_LEN * H_DIM);

  k_combine<<<4096, 256, 0, stream>>>(outG, outL, gate, att);
  k_pool<<<B_SZ, 512, 0, stream>>>(att, pooled);
  k_bnfc<<<1, 1024, 0, stream>>>(pooled, bng, bnb, Wfc, bfc, (float*)d_out);
}
